// skip_torchlstm_cell_80513456930853
// MI455X (gfx1250) — matrix operand order repaired and checked by reading, not compiled, not run on hardware
//
#include <hip/hip_runtime.h>

// ---------------------------------------------------------------------------
// Types for CDNA5 WMMA (wave32)
// ---------------------------------------------------------------------------
typedef __bf16  v16bf __attribute__((ext_vector_type(16)));
typedef float   v8f   __attribute__((ext_vector_type(8)));

// Problem constants
#define BATCH 16384
#define HID   512
#define KDIM  1024      // concat K for both GEMMs
#define N1    2048      // 4*H gates
#define N2    512       // sg(256) + tg(256)

// GEMM tiling
#define TILE_M 128
#define TILE_N 128
#define TILE_K 32
#define LDSTR  40       // padded LDS row stride in bf16 elems (80B -> 20 banks)

__device__ __forceinline__ __bf16 f2bf(float f) {
    // round-to-nearest-even fp32 -> bf16
    unsigned u = __builtin_bit_cast(unsigned, f);
    unsigned r = (u + 0x7FFFu + ((u >> 16) & 1u)) >> 16;
    unsigned short s = (unsigned short)r;
    return __builtin_bit_cast(__bf16, s);
}

__device__ __forceinline__ float sigmoidf_(float x) {
    return 1.0f / (1.0f + __expf(-x));
}

// ---------------------------------------------------------------------------
// CDNA5 async global->LDS copy (ASYNCcnt-tracked, no VGPR staging).
// lds_off: byte offset of destination within the workgroup LDS allocation
// (low 32 bits of a generic pointer to __shared__ memory).
// ---------------------------------------------------------------------------
__device__ __forceinline__ void async_copy_b128(void* lds_dst, const void* gsrc) {
    unsigned lds_off = (unsigned)(uintptr_t)lds_dst;
    asm volatile("global_load_async_to_lds_b128 %0, %1, off"
                 :
                 : "v"(lds_off), "v"(gsrc)
                 : "memory");
}

__device__ __forceinline__ void wait_async0() {
#if __has_builtin(__builtin_amdgcn_s_wait_asynccnt)
    __builtin_amdgcn_s_wait_asynccnt(0);
#else
    asm volatile("s_wait_asynccnt 0x0" ::: "memory");
#endif
}

// ---------------------------------------------------------------------------
// Pack two fp32 [rows, cols] matrices side-by-side into bf16 [rows, 2*cols].
// srcR == nullptr -> zero-fill the right half.
// ---------------------------------------------------------------------------
__global__ __launch_bounds__(256)
void pack_pair_bf16(__bf16* __restrict__ dst,
                    const float* __restrict__ srcL,
                    const float* __restrict__ srcR,
                    int rows, int cols) {
    int idx = blockIdx.x * blockDim.x + threadIdx.x;
    int total = rows * cols;
    if (idx >= total) return;
    int r = idx / cols;
    int c = idx - r * cols;
    __bf16* drow = dst + (size_t)r * (2 * cols);
    drow[c]        = f2bf(srcL[idx]);
    drow[cols + c] = f2bf(srcR ? srcR[idx] : 0.0f);
}

// ---------------------------------------------------------------------------
// C[M,N] (f32) = A[M,K] (bf16, K-major) * B[N,K]^T (bf16, K-major)
// Block: 256 threads = 8 waves, 4x2 wave grid, wave tile 32x64 (2x4 WMMA).
// Double-buffered LDS filled by global_load_async_to_lds_b128.
// N, K compile-time so epilogue/staging strides become immediates.
// ---------------------------------------------------------------------------
template<int N, int K>
__global__ __launch_bounds__(256)
void gemm_bf16_wmma(const __bf16* __restrict__ A,
                    const __bf16* __restrict__ Bw,
                    float* __restrict__ C) {
    __shared__ __attribute__((aligned(16))) __bf16 lA[2][TILE_M * LDSTR];
    __shared__ __attribute__((aligned(16))) __bf16 lB[2][TILE_N * LDSTR];

    const int t    = threadIdx.x;
    const int lane = t & 31;
    const int w    = t >> 5;        // wave 0..7
    const int wm   = w >> 1;        // 0..3  -> 32-row strips
    const int wn   = w & 1;         // 0..1  -> 64-col strips
    const int h    = lane >> 4;     // lane half (ISA bf16 frag layout)
    const int lm   = lane & 15;

    const int m0 = blockIdx.y * TILE_M;
    const int n0 = blockIdx.x * TILE_N;

    // this thread's two staging chunks (16B each) per tile
    const int sc0  = t;              // chunk ids 0..511
    const int sc1  = t + 256;
    const int row0 = sc0 >> 2, c16_0 = (sc0 & 3) * 8;
    const int row1 = sc1 >> 2, c16_1 = (sc1 & 3) * 8;

    v8f acc[2][4];
#pragma unroll
    for (int i = 0; i < 2; ++i)
#pragma unroll
        for (int j = 0; j < 4; ++j)
#pragma unroll
            for (int v = 0; v < 8; ++v) acc[i][j][v] = 0.0f;

    union FragU { uint4 u[2]; v16bf v; };

    // prologue: fill buffer 0
    {
        const int kt = 0;
        async_copy_b128(&lA[0][row0 * LDSTR + c16_0],
                        A + (size_t)(m0 + row0) * K + kt + c16_0);
        async_copy_b128(&lB[0][row0 * LDSTR + c16_0],
                        Bw + (size_t)(n0 + row0) * K + kt + c16_0);
        async_copy_b128(&lA[0][row1 * LDSTR + c16_1],
                        A + (size_t)(m0 + row1) * K + kt + c16_1);
        async_copy_b128(&lB[0][row1 * LDSTR + c16_1],
                        Bw + (size_t)(n0 + row1) * K + kt + c16_1);
    }
    wait_async0();
    __syncthreads();

    constexpr int NK = K / TILE_K;
    for (int ki = 0; ki < NK; ++ki) {
        const int buf = ki & 1;

        // prefetch next slab into the other buffer while we compute
        if (ki + 1 < NK) {
            const int kt = (ki + 1) * TILE_K;
            async_copy_b128(&lA[buf ^ 1][row0 * LDSTR + c16_0],
                            A + (size_t)(m0 + row0) * K + kt + c16_0);
            async_copy_b128(&lB[buf ^ 1][row0 * LDSTR + c16_0],
                            Bw + (size_t)(n0 + row0) * K + kt + c16_0);
            async_copy_b128(&lA[buf ^ 1][row1 * LDSTR + c16_1],
                            A + (size_t)(m0 + row1) * K + kt + c16_1);
            async_copy_b128(&lB[buf ^ 1][row1 * LDSTR + c16_1],
                            Bw + (size_t)(n0 + row1) * K + kt + c16_1);
        }

        // A fragments: 16x32 bf16; lane half h holds K chunks [h*8..) and [16+h*8..)
        v16bf af[2];
#pragma unroll
        for (int i = 0; i < 2; ++i) {
            const __bf16* p = &lA[buf][(wm * 32 + i * 16 + lm) * LDSTR];
            FragU f;
            f.u[0] = *(const uint4*)(p + h * 8);
            f.u[1] = *(const uint4*)(p + 16 + h * 8);
            af[i] = f.v;
        }
        // B fragments: 32x16 (KxN); lane half h holds K chunks [h*8..) and [16+h*8..)
        v16bf bfr[4];
#pragma unroll
        for (int j = 0; j < 4; ++j) {
            const __bf16* p = &lB[buf][(wn * 64 + j * 16 + lm) * LDSTR];
            FragU f;
            f.u[0] = *(const uint4*)(p + h * 8);
            f.u[1] = *(const uint4*)(p + 16 + h * 8);
            bfr[j] = f.v;
        }

#pragma unroll
        for (int i = 0; i < 2; ++i)
#pragma unroll
            for (int j = 0; j < 4; ++j)
                acc[i][j] = __builtin_amdgcn_wmma_f32_16x16x32_bf16(
                    false, af[i], false, bfr[j], (short)0, acc[i][j],
                    false, false);

        // next buffer fully landed + everyone done reading current buffer
        wait_async0();
        __syncthreads();
    }

    // C/D layout: VGPR v -> row (h*8 + v), col = lm; strides are immediates.
#pragma unroll
    for (int i = 0; i < 2; ++i) {
        const int mrow = m0 + wm * 32 + i * 16 + h * 8;
        float* cp = C + (size_t)mrow * N + n0 + wn * 64 + lm;
#pragma unroll
        for (int j = 0; j < 4; ++j)
#pragma unroll
            for (int v = 0; v < 8; ++v)
                cp[j * 16 + (size_t)v * N] = acc[i][j][v];
    }
}

// ---------------------------------------------------------------------------
// LSTM elementwise: gates[B,2048] (+biases) -> c1,h1 (f32) and A2=[c0|c1] bf16
// ---------------------------------------------------------------------------
__global__ __launch_bounds__(256)
void lstm_elem(const float* __restrict__ gates,
               const float* __restrict__ c0,
               const float* __restrict__ b_ih,
               const float* __restrict__ b_hh,
               float* __restrict__ h1,
               float* __restrict__ c1,
               __bf16* __restrict__ A2) {
    int idx = blockIdx.x * blockDim.x + threadIdx.x;   // 0 .. B*H
    if (idx >= BATCH * HID) return;
    int b = idx >> 9;
    int hc = idx & (HID - 1);
    const float* gr = gates + (size_t)b * N1;
    float gi = gr[hc]            + b_ih[hc]            + b_hh[hc];
    float gf = gr[HID + hc]      + b_ih[HID + hc]      + b_hh[HID + hc];
    float gg = gr[2 * HID + hc]  + b_ih[2 * HID + hc]  + b_hh[2 * HID + hc];
    float go = gr[3 * HID + hc]  + b_ih[3 * HID + hc]  + b_hh[3 * HID + hc];
    float ig = sigmoidf_(gi);
    float fg = sigmoidf_(gf);
    float g  = tanhf(gg);
    float og = sigmoidf_(go);
    float c0v = c0[idx];
    float c1v = fg * c0v + ig * g;
    float h1v = og * tanhf(c1v);
    c1[idx] = c1v;
    h1[idx] = h1v;
    __bf16* ar = A2 + (size_t)b * KDIM;
    ar[hc]       = f2bf(c0v);
    ar[HID + hc] = f2bf(c1v);
}

// ---------------------------------------------------------------------------
// Per-row finalize: leaky-relu + dot reductions + sigmoid gates + blend.
// One block (256 threads) per batch row.
// ---------------------------------------------------------------------------
__global__ __launch_bounds__(256)
void finalize(const float* __restrict__ sgtg,
              const float* __restrict__ b_sp, const float* __restrict__ b_sc,
              const float* __restrict__ W_so, const float* __restrict__ b_so,
              const float* __restrict__ b_tp,
              const float* __restrict__ W_to, const float* __restrict__ b_to,
              const float* __restrict__ cum,
              const float* __restrict__ h0, const float* __restrict__ c0,
              const float* __restrict__ h1, const float* __restrict__ c1,
              float* __restrict__ out_h, float* __restrict__ out_c,
              float* __restrict__ out_cum, float* __restrict__ out_delta,
              float* __restrict__ out_prob) {
    __shared__ float red1[256];
    __shared__ float red2[256];
    __shared__ float s_gate;
    const int b = blockIdx.x;
    const int t = threadIdx.x;

    const float* row = sgtg + (size_t)b * N2;
    float sg = row[t] + b_sp[t] + b_sc[t];
    sg = sg > 0.0f ? sg : 0.01f * sg;
    float tg = row[256 + t] + b_tp[t];
    tg = tg > 0.0f ? tg : 0.01f * tg;
    red1[t] = sg * W_so[t];
    red2[t] = tg * W_to[t];
    __syncthreads();
#pragma unroll
    for (int s = 128; s > 0; s >>= 1) {
        if (t < s) { red1[t] += red1[t + s]; red2[t] += red2[t + s]; }
        __syncthreads();
    }
    if (t == 0) {
        float delta = sigmoidf_(red1[0] + b_so[0]);
        float thr   = sigmoidf_(red2[0] + b_to[0]);
        float cu    = cum[b];
        float rem   = 1.0f - cu;
        float prob  = cu + (delta < rem ? delta : rem);
        float gate  = prob > thr ? 1.0f : 0.0f;   // straight-through fwd value
        s_gate      = gate;
        out_cum[b]   = (1.0f - gate) * prob;
        out_delta[b] = delta;
        out_prob[b]  = prob;
    }
    __syncthreads();
    const float gate = s_gate;
    const size_t base = (size_t)b * HID;
#pragma unroll
    for (int c = t; c < HID; c += 256) {
        out_h[base + c] = gate * h1[base + c] + (1.0f - gate) * h0[base + c];
        out_c[base + c] = gate * c1[base + c] + (1.0f - gate) * c0[base + c];
    }
}

// ---------------------------------------------------------------------------
// Host launcher
// ---------------------------------------------------------------------------
extern "C" void kernel_launch(void* const* d_in, const int* in_sizes, int n_in,
                              void* d_out, int out_size, void* d_ws, size_t ws_size,
                              hipStream_t stream) {
    (void)in_sizes; (void)n_in; (void)out_size; (void)ws_size;
    // inputs (setup_inputs order)
    const float* x    = (const float*)d_in[0];
    const float* h0   = (const float*)d_in[1];
    const float* c0   = (const float*)d_in[2];
    const float* cum  = (const float*)d_in[3];
    const float* W_ih = (const float*)d_in[4];
    const float* W_hh = (const float*)d_in[5];
    const float* b_ih = (const float*)d_in[6];
    const float* b_hh = (const float*)d_in[7];
    const float* W_sp = (const float*)d_in[8];
    const float* b_sp = (const float*)d_in[9];
    const float* W_sc = (const float*)d_in[10];
    const float* b_sc = (const float*)d_in[11];
    const float* W_so = (const float*)d_in[12];
    const float* b_so = (const float*)d_in[13];
    const float* W_tp = (const float*)d_in[14];
    const float* b_tp = (const float*)d_in[15];
    const float* W_to = (const float*)d_in[16];
    const float* b_to = (const float*)d_in[17];

    // workspace layout (bytes)
    char* ws = (char*)d_ws;
    __bf16* A1    = (__bf16*)(ws);                              // 32 MB  [B,1024]
    __bf16* W1    = (__bf16*)(ws + 33554432);                   //  4 MB  [2048,1024]
    __bf16* W2    = (__bf16*)(ws + 37748736);                   //  1 MB  [512,1024]
    float*  gates = (float* )(ws + 38797312);                   // 128 MB [B,2048]
    float*  h1    = (float* )(ws + 173015040);                  // 32 MB
    float*  c1    = (float* )(ws + 206569472);                  // 32 MB
    __bf16* A2    = (__bf16*)(ws + 240123904);                  // 32 MB  [B,1024]
    float*  sgtg  = gates;   // overlay: gates dead after lstm_elem; [B,512]

    // outputs: new_h, new_c, update_cum, update_prob_delta, update_prob
    float* out_h     = (float*)d_out;
    float* out_c     = out_h + (size_t)BATCH * HID;
    float* out_cum   = out_c + (size_t)BATCH * HID;
    float* out_delta = out_cum + BATCH;
    float* out_prob  = out_delta + BATCH;

    const int TB = 256;

    // 1. pack bf16 operands
    pack_pair_bf16<<<(BATCH * HID + TB - 1) / TB, TB, 0, stream>>>(
        A1, x, h0, BATCH, HID);
    pack_pair_bf16<<<(N1 * HID + TB - 1) / TB, TB, 0, stream>>>(
        W1, W_ih, W_hh, N1, HID);
    pack_pair_bf16<<<(256 * HID + TB - 1) / TB, TB, 0, stream>>>(
        W2, W_sp, W_sc, 256, HID);
    pack_pair_bf16<<<(256 * HID + TB - 1) / TB, TB, 0, stream>>>(
        W2 + (size_t)256 * KDIM, W_tp, (const float*)nullptr, 256, HID);

    // 2. big GEMM: gates = [x|h0] * [W_ih|W_hh]^T
    gemm_bf16_wmma<N1, KDIM>
        <<<dim3(N1 / TILE_N, BATCH / TILE_M), TB, 0, stream>>>(A1, W1, gates);

    // 3. LSTM nonlinearity -> c1, h1, A2=[c0|c1] bf16
    lstm_elem<<<(BATCH * HID + TB - 1) / TB, TB, 0, stream>>>(
        gates, c0, b_ih, b_hh, h1, c1, A2);

    // 4. MLP hidden GEMM: sgtg = [c0|c1] * [[W_sp|W_sc];[W_tp|0]]^T
    gemm_bf16_wmma<N2, KDIM>
        <<<dim3(N2 / TILE_N, BATCH / TILE_M), TB, 0, stream>>>(A2, W2, sgtg);

    // 5. reductions, gating, blend
    finalize<<<BATCH, TB, 0, stream>>>(
        sgtg, b_sp, b_sc, W_so, b_so, b_tp, W_to, b_to,
        cum, h0, c0, h1, c1,
        out_h, out_c, out_cum, out_delta, out_prob);
}
